// MultiHeadAttention_18803366822348
// MI455X (gfx1250) — matrix operand order repaired and checked by reading, not compiled, not run on hardware
//
#include <hip/hip_runtime.h>
#include <hip/hip_bf16.h>

typedef __bf16 bf16;
typedef bf16  v16bf __attribute__((ext_vector_type(16)));
typedef bf16  v8bf  __attribute__((ext_vector_type(8)));
typedef bf16  v4bf  __attribute__((ext_vector_type(4)));
typedef float v8f   __attribute__((ext_vector_type(8)));
typedef float v4f   __attribute__((ext_vector_type(4)));
typedef unsigned int v4u __attribute__((ext_vector_type(4)));

#define D_MODEL   2048
#define N_HEADS   16
#define HEAD_DIM  128
#define SEQ       2048
#define BATCH     2
#define MROWS     (BATCH*SEQ)      // 4096
#define NQKV      (3*D_MODEL)      // 6144

__device__ __forceinline__ v16bf join8(v8bf lo, v8bf hi) {
  return __builtin_shufflevector(lo, hi, 0,1,2,3,4,5,6,7,8,9,10,11,12,13,14,15);
}
__device__ __forceinline__ v8f wmma_bf16(v16bf a, v16bf b, v8f c) {
  // D = A(16x32 bf16) x B(32x16 bf16) + C(16x16 f32)
  return __builtin_amdgcn_wmma_f32_16x16x32_bf16(false, a, false, b, (short)0, c, false, false);
}
// CDNA5 async global->LDS copy (16B per lane), tracked by ASYNCcnt.
__device__ __forceinline__ void async_g2l_b128(const bf16* gptr, bf16* ldsptr) {
  unsigned l = (unsigned)(size_t)ldsptr;   // flat LDS addr truncates to LDS offset
  asm volatile("global_load_async_to_lds_b128 %0, %1, off"
               :: "v"(l), "v"(gptr) : "memory");
}
__device__ __forceinline__ void wait_async0() {
  asm volatile("s_wait_asynccnt 0" ::: "memory");
}

// ---------------------------------------------------------------- convert (4-wide)
__global__ void cvt_f32_bf16(const float* __restrict__ src, bf16* __restrict__ dst, int n4) {
  int i = blockIdx.x * blockDim.x + threadIdx.x;
  if (i < n4) {
    v4f v = ((const v4f*)src)[i];
    ((v4bf*)dst)[i] = __builtin_convertvector(v, v4bf);
  }
}

// ---------------------------------------------------------------- GEMM  out = A(MxK) * B(NxK)^T
// MODE 0: plain, f32 out [M][N].  MODE 1: QKV + fused RoPE, bf16 out to Q/K/V [B*H][SEQ][128].
template<int MODE>
__global__ __launch_bounds__(256)
void gemm128(const bf16* __restrict__ A, const bf16* __restrict__ B,
             float* __restrict__ Cout,
             bf16* __restrict__ Qb, bf16* __restrict__ Kb, bf16* __restrict__ Vb,
             int N, int Kd)
{
  __shared__ bf16 As[128 * 48];   // [row][k] pad to 48 elems (96B rows, 16B aligned)
  __shared__ bf16 Bs[128 * 48];

  const int tid   = threadIdx.x;
  const int wave  = tid >> 5;
  const int lane  = tid & 31;
  const int lhalf = lane >> 4;    // 0|1
  const int l16   = lane & 15;

  const long mbase = (long)blockIdx.y * 128;
  const long nbase = (long)blockIdx.x * 128;

  v8f acc[8];
  #pragma unroll
  for (int f = 0; f < 8; ++f) acc[f] = (v8f){0.f,0.f,0.f,0.f,0.f,0.f,0.f,0.f};

  for (int kt = 0; kt < Kd; kt += 32) {
    __syncthreads();
    #pragma unroll
    for (int it = 0; it < 2; ++it) {              // 512 16B chunks each / 256 threads
      int chunk = tid + it * 256;
      int row = chunk >> 2;
      int cid = chunk & 3;
      async_g2l_b128(A + (mbase + row) * Kd + kt + cid * 8, As + row * 48 + cid * 8);
      async_g2l_b128(B + (nbase + row) * Kd + kt + cid * 8, Bs + row * 48 + cid * 8);
    }
    if (kt + 32 < Kd) {                           // gfx1250 global_prefetch_b8
      __builtin_prefetch(A + (mbase + (tid >> 1)) * Kd + kt + 32, 0, 1);
      __builtin_prefetch(B + (nbase + (tid >> 1)) * Kd + kt + 32, 0, 1);
    }
    wait_async0();
    __syncthreads();

    // A fragment: lane = row (wave's 16-row strip), K chunks {0-7,16-23} / {8-15,24-31}
    const bf16* ap = As + (wave * 16 + l16) * 48 + lhalf * 8;
    v16bf afrag = join8(*(const v8bf*)ap, *(const v8bf*)(ap + 16));
    // preload all 8 B fragments, then issue WMMAs back-to-back
    v16bf bfrag[8];
    #pragma unroll
    for (int f = 0; f < 8; ++f) {
      const bf16* bp = Bs + (f * 16 + l16) * 48 + lhalf * 8;    // lane = col, K chunks {0-7,16-23} / {8-15,24-31}
      bfrag[f] = join8(*(const v8bf*)bp, *(const v8bf*)(bp + 16));
    }
    #pragma unroll
    for (int f = 0; f < 8; ++f)
      acc[f] = wmma_bf16(afrag, bfrag[f], acc[f]);
  }

  if (MODE == 0) {
    #pragma unroll
    for (int f = 0; f < 8; ++f)
      #pragma unroll
      for (int r = 0; r < 8; ++r) {
        long row = mbase + wave * 16 + r + lhalf * 8;
        long col = nbase + f * 16 + l16;
        Cout[row * N + col] = acc[f][r];
      }
  } else {
    const int ncol0 = (int)nbase;
    const int which = ncol0 >> 11;          // 0=Q 1=K 2=V
    const int h     = (ncol0 >> 7) & (N_HEADS - 1);
    bf16* dst = (which == 0) ? Qb : ((which == 1) ? Kb : Vb);
    if (which < 2) {                        // fused RoPE: pair d and d+64 = frag f and f+4
      #pragma unroll
      for (int f = 0; f < 4; ++f) {
        float j   = (float)(f * 16 + l16);          // 0..63
        float inv = __expf(j * -0.14391156516f);    // theta^(-j/64), ln(1e4)/64
        #pragma unroll
        for (int r = 0; r < 8; ++r) {
          int m = (int)mbase + wave * 16 + r + lhalf * 8;
          int s = m & (SEQ - 1);
          float ang = (float)s * inv;
          float sn, cs;
          __sincosf(ang, &sn, &cs);
          float x1 = acc[f][r], x2 = acc[f + 4][r];
          acc[f][r]     = x1 * cs - x2 * sn;
          acc[f + 4][r] = x2 * cs + x1 * sn;
        }
      }
    }
    #pragma unroll
    for (int f = 0; f < 8; ++f)
      #pragma unroll
      for (int r = 0; r < 8; ++r) {
        int m = (int)mbase + wave * 16 + r + lhalf * 8;
        int s = m & (SEQ - 1), b = m >> 11;
        int d = f * 16 + l16;
        dst[(((long)(b * N_HEADS + h) * SEQ + s) << 7) + d] = (bf16)acc[f][r];
      }
  }
}

// ---------------------------------------------------------------- flash attention
// grid: (SEQ/64, B*H), 128 threads = 4 waves, 16 q-rows per wave, K-tile 32.
__global__ __launch_bounds__(128)
void attn_kernel(const bf16* __restrict__ Qb, const bf16* __restrict__ Kb,
                 const bf16* __restrict__ Vb, bf16* __restrict__ ctx)
{
  __shared__ bf16 Ks[32 * 136];      // [k][d]  row stride 272B (16B aligned)
  __shared__ bf16 Vt[128 * 40];      // [d][k]  row stride 80B  (16B aligned)
  __shared__ bf16 Plds[4 * 16 * 32]; // per-wave P transpose scratch

  const int tid   = threadIdx.x;
  const int wave  = tid >> 5;
  const int lane  = tid & 31;
  const int lhalf = lane >> 4;
  const int l16   = lane & 15;
  const int bh    = blockIdx.y;
  const int q0    = blockIdx.x * 64;

  // preload Q fragments (16 q-rows x 128d) straight from global in A-layout
  const bf16* qp = Qb + ((long)bh * SEQ + q0 + wave * 16 + l16) * HEAD_DIM;
  v16bf qf[4];
  #pragma unroll
  for (int c = 0; c < 4; ++c) {
    const bf16* p = qp + c * 32 + lhalf * 8;
    qf[c] = join8(*(const v8bf*)p, *(const v8bf*)(p + 16));
  }

  float mrow[8], lrow[8];
  v8f o[8];
  #pragma unroll
  for (int r = 0; r < 8; ++r) { mrow[r] = -1e30f; lrow[r] = 0.f; }
  #pragma unroll
  for (int f = 0; f < 8; ++f) o[f] = (v8f){0.f,0.f,0.f,0.f,0.f,0.f,0.f,0.f};

  const float scale = 0.088388347648318f;  // 1/sqrt(128)
  const long  kvbase = (long)bh * SEQ * HEAD_DIM;
  const int   kend = q0 + 64;

  for (int kt = 0; kt < kend; kt += 32) {
    __syncthreads();
    #pragma unroll
    for (int it = 0; it < 4; ++it) {            // async-stage K tile [32][128]
      int chunk = tid + it * 128;
      int krow = chunk >> 4, cid = chunk & 15;
      async_g2l_b128(Kb + kvbase + (long)(kt + krow) * HEAD_DIM + cid * 8,
                     Ks + krow * 136 + cid * 8);
    }
    for (int idx = tid; idx < 32 * 128; idx += 128) {   // stage V transposed [d][k]
      int k = idx >> 7, d = idx & 127;
      Vt[d * 40 + k] = Vb[kvbase + (long)(kt + k) * HEAD_DIM + d];
    }
    wait_async0();
    __syncthreads();

    // preload 8 K fragments (4 d-chunks x 2 score tiles), then batch the WMMAs
    v16bf kf[8];
    #pragma unroll
    for (int c = 0; c < 4; ++c) {
      const bf16* k0 = Ks + l16 * 136        + c * 32 + lhalf * 8;
      const bf16* k1 = Ks + (16 + l16) * 136 + c * 32 + lhalf * 8;
      kf[c]     = join8(*(const v8bf*)k0, *(const v8bf*)(k0 + 16));
      kf[4 + c] = join8(*(const v8bf*)k1, *(const v8bf*)(k1 + 16));
    }
    v8f s0 = (v8f){0.f,0.f,0.f,0.f,0.f,0.f,0.f,0.f};
    v8f s1 = s0;
    #pragma unroll
    for (int c = 0; c < 4; ++c) {
      s0 = wmma_bf16(qf[c], kf[c],     s0);
      s1 = wmma_bf16(qf[c], kf[4 + c], s1);
    }

    const int col0 = kt + l16, col1 = kt + 16 + l16;
    #pragma unroll
    for (int r = 0; r < 8; ++r) {
      int row = q0 + wave * 16 + r + lhalf * 8;
      float v0 = s0[r] * scale; if (col0 > row) v0 = -1e30f;
      float v1 = s1[r] * scale; if (col1 > row) v1 = -1e30f;
      // online softmax (rows live across the 16 lanes of each half-wave)
      float mx = fmaxf(v0, v1);
      #pragma unroll
      for (int off = 8; off; off >>= 1) mx = fmaxf(mx, __shfl_xor(mx, off, 16));
      float nm = fmaxf(mrow[r], mx);
      float alpha = __expf(mrow[r] - nm);
      float p0 = __expf(v0 - nm);
      float p1 = __expf(v1 - nm);
      float rs = p0 + p1;
      #pragma unroll
      for (int off = 8; off; off >>= 1) rs += __shfl_xor(rs, off, 16);
      lrow[r] = lrow[r] * alpha + rs;
      mrow[r] = nm;
      s0[r] = p0; s1[r] = p1;
      #pragma unroll
      for (int f = 0; f < 8; ++f) o[f][r] *= alpha;
    }

    // transpose P (C-layout -> A-layout) through per-wave LDS; same-wave DS is in-order
    bf16* pl = Plds + wave * (16 * 32);
    #pragma unroll
    for (int r = 0; r < 8; ++r) {
      int row = r + lhalf * 8;
      pl[row * 32 + l16]      = (bf16)s0[r];
      pl[row * 32 + 16 + l16] = (bf16)s1[r];
    }
    const bf16* pa = pl + l16 * 32 + lhalf * 8;
    v16bf pfrag = join8(*(const v8bf*)pa, *(const v8bf*)(pa + 16));

    // preload 8 V fragments, then batch the WMMAs
    v16bf vf[8];
    #pragma unroll
    for (int f = 0; f < 8; ++f) {
      const bf16* vp = Vt + (f * 16 + l16) * 40 + lhalf * 8;
      vf[f] = join8(*(const v8bf*)vp, *(const v8bf*)(vp + 16));
    }
    #pragma unroll
    for (int f = 0; f < 8; ++f)                 // O += P(16x32) x V(32x16) per d-chunk
      o[f] = wmma_bf16(pfrag, vf[f], o[f]);
  }

  const int b = bh >> 4, h = bh & 15;
  #pragma unroll
  for (int f = 0; f < 8; ++f)
    #pragma unroll
    for (int r = 0; r < 8; ++r) {
      int s = q0 + wave * 16 + r + lhalf * 8;
      float val = o[f][r] / lrow[r];
      ctx[((long)b * SEQ + s) * D_MODEL + h * HEAD_DIM + f * 16 + l16] = (bf16)val;
    }
}

// ---------------------------------------------------------------- launch
extern "C" void kernel_launch(void* const* d_in, const int* in_sizes, int n_in,
                              void* d_out, int out_size, void* d_ws, size_t ws_size,
                              hipStream_t stream) {
  const float* x  = (const float*)d_in[0];
  const float* Wq = (const float*)d_in[1];
  const float* Wk = (const float*)d_in[2];
  const float* Wv = (const float*)d_in[3];
  const float* Wo = (const float*)d_in[4];
  float* out = (float*)d_out;

  char* ws = (char*)d_ws;
  size_t off = 0;
  bf16* xb    = (bf16*)(ws + off); off += (size_t)MROWS * D_MODEL * 2;      // 16 MB
  bf16* Wqkvb = (bf16*)(ws + off); off += (size_t)NQKV * D_MODEL * 2;       // 24 MB
  bf16* Wob   = (bf16*)(ws + off); off += (size_t)D_MODEL * D_MODEL * 2;    //  8 MB
  bf16* Qb    = (bf16*)(ws + off); off += (size_t)MROWS * D_MODEL * 2;      // 16 MB
  bf16* Kb    = (bf16*)(ws + off); off += (size_t)MROWS * D_MODEL * 2;      // 16 MB
  bf16* Vb    = (bf16*)(ws + off); off += (size_t)MROWS * D_MODEL * 2;      // 16 MB
  bf16* ctxb  = (bf16*)(ws + off); off += (size_t)MROWS * D_MODEL * 2;      // 16 MB
  (void)ws_size; (void)in_sizes; (void)n_in; (void)out_size;

  const int W2 = D_MODEL * D_MODEL;
  cvt_f32_bf16<<<(MROWS * D_MODEL / 4) / 256, 256, 0, stream>>>(x, xb, MROWS * D_MODEL / 4);
  cvt_f32_bf16<<<(W2 / 4) / 256, 256, 0, stream>>>(Wq, Wqkvb,              W2 / 4);
  cvt_f32_bf16<<<(W2 / 4) / 256, 256, 0, stream>>>(Wk, Wqkvb + (size_t)W2,     W2 / 4);
  cvt_f32_bf16<<<(W2 / 4) / 256, 256, 0, stream>>>(Wv, Wqkvb + (size_t)2 * W2, W2 / 4);
  cvt_f32_bf16<<<(W2 / 4) / 256, 256, 0, stream>>>(Wo, Wob, W2 / 4);

  // QKV projection + fused RoPE -> bf16 [B*H][SEQ][128]
  gemm128<1><<<dim3(NQKV / 128, MROWS / 128), 256, 0, stream>>>(
      xb, Wqkvb, nullptr, Qb, Kb, Vb, NQKV, D_MODEL);

  // causal flash attention -> ctx bf16 [B][SEQ][D_MODEL]
  attn_kernel<<<dim3(SEQ / 64, BATCH * N_HEADS), 128, 0, stream>>>(Qb, Kb, Vb, ctxb);

  // out = ctx @ Wo^T, f32
  gemm128<0><<<dim3(D_MODEL / 128, MROWS / 128), 256, 0, stream>>>(
      ctxb, Wob, out, nullptr, nullptr, nullptr, D_MODEL, D_MODEL);
}
